// NeuralTensorLayer_28604482191498
// MI455X (gfx1250) — matrix operand order repaired and checked by reading, not compiled, not run on hardware
//
#include <hip/hip_runtime.h>
#include <hip/hip_bf16.h>
#include <math.h>

#define BSZ 131072
#define DD  64
#define KBI 4096           // bilinear K = 64*64
#define KFF 128            // feed-forward K = 2*64
#define KTOT (KBI + KFF)   // 4224

typedef __attribute__((ext_vector_type(16))) __bf16 v16bf;
typedef __attribute__((ext_vector_type(8)))  __bf16 v8bf;
typedef __attribute__((ext_vector_type(8)))  float  v8f;

// ---------------------------------------------------------------------------
// Prep: build 3 bf16 weight matrices Tt[t] of shape (64, 4224) row-major:
//   cols [0,4096)  = T_t[k].flat  (i*64+j)
//   cols [4096,..) = W[k, 0..127]
// ---------------------------------------------------------------------------
__global__ void prep_weights(const float* __restrict__ T1,
                             const float* __restrict__ T2,
                             const float* __restrict__ T3,
                             const float* __restrict__ W,
                             __bf16* __restrict__ out) {
    int idx = blockIdx.x * blockDim.x + threadIdx.x;
    const int total = 3 * DD * KTOT;
    if (idx >= total) return;
    int t   = idx / (DD * KTOT);
    int rem = idx - t * (DD * KTOT);
    int k   = rem / KTOT;
    int c   = rem - k * KTOT;
    const float* T = (t == 0) ? T1 : ((t == 1) ? T2 : T3);
    float v = (c < KBI) ? T[(size_t)k * KBI + c] : W[k * KFF + (c - KBI)];
    out[idx] = (__bf16)v;
}

// A fragment: 16x32 bf16, lane holds row = lane&15, K = kx + e + half*16
static __device__ __forceinline__ v16bf load_afrag(const __bf16* __restrict__ Arow,
                                                   int kx, int half) {
    v8bf lo = *(const v8bf*)(Arow + kx + half * 16);
    v8bf hi = *(const v8bf*)(Arow + kx + half * 16 + 8);
    return __builtin_shufflevector(lo, hi, 0, 1, 2, 3, 4, 5, 6, 7,
                                           8, 9, 10, 11, 12, 13, 14, 15);
}

#define WMMA_BF16(A, B, C) \
    __builtin_amdgcn_wmma_f32_16x16x32_bf16(false, (A), false, (B), (short)0, (C), false, false)

// ---------------------------------------------------------------------------
// Fused NTN stage:  X(64, BSZ) = tanh( Tt(64,4224) @ Btot(4224,BSZ) + bias )
//   Btot[ij, b]     = E1[b, ij>>6] * E2[b, ij&63]          (ij   < 4096)
//   Btot[4096+r, s] = E[r*1024 + (s>>7), s&127]            (ffp gather)
// Block = 128 threads (4 waves), 64-column batch tile per block.
// Wave w owns batch columns [16w,16w+16) and computes ALL 64 output rows
// (4 accumulators) so each B fragment is amortized over 4 WMMAs.
// e2 row is preloaded into registers, half-rotated so fragment indices are
// compile-time constants; inner loop has 1 ds_load per 64 K.
// ---------------------------------------------------------------------------
__global__ __launch_bounds__(128) void ntn_stage(
    const float* __restrict__ E1,    // (BSZ, 64)
    const float* __restrict__ E2,    // (BSZ, 64)
    const __bf16* __restrict__ Tt,   // (64, 4224) bf16
    const float* __restrict__ bias,  // (64,)
    float* __restrict__ X)           // (64, BSZ)
{
    __shared__ __attribute__((aligned(16))) float  e1s[64][68];   // 272B stride: aligned + bank-spread
    __shared__ __attribute__((aligned(16))) float  e2s[64][68];
    __shared__ __attribute__((aligned(16))) __bf16 gsb[64][136];  // G col-major, bf16

    const int tid  = threadIdx.x;
    const int wave = tid >> 5;
    const int lane = tid & 31;
    const int half = lane >> 4;      // 0: lanes 0-15, 1: lanes 16-31
    const int colw = lane & 15;      // WMMA N column / A-matrix M row (lane&15)
    const int colb = wave * 16 + colw;   // batch column within 64-wide tile
    const int s0   = blockIdx.x * 64;

    // ---- stage e1/e2 tile rows (threads 0-63: e1 row tid; 64-127: e2 row tid-64)
    {
        int r = tid & 63;
        const float* src = (tid < 64) ? (E1 + (size_t)(s0 + r) * DD)
                                      : (E2 + (size_t)(s0 + r) * DD);
        float* dst = (tid < 64) ? &e1s[r][0] : &e2s[r][0];
#pragma unroll
        for (int q = 0; q < 16; ++q) {
            float4 v = ((const float4*)src)[q];
            *(float4*)(dst + q * 4) = v;
        }
    }
    // ---- stage G (ffp gather) in bf16, col-major: gsb[col][r] = E[r*1024+bb, c0+col]
    // s0 is 64-aligned => c0 = s0&127 in {0,64} => source column == gcol in e1 OR e2.
    {
        int bb = s0 >> 7;
        const float* Esrc = ((s0 & 127) < DD) ? E1 : E2;
        int gcol = tid & 63;
        int r0 = (tid >> 6) * 64;    // 0 or 64
#pragma unroll
        for (int q = 0; q < 64; ++q) {
            int r = r0 + q;
            gsb[gcol][r] = (__bf16)Esrc[(size_t)(r * 1024 + bb) * DD + gcol];
        }
    }
    __syncthreads();

    // ---- preload e2 row into registers, rotated by half*16:
    //      yb[t] = e2[s0+colb, (t + half*16) & 63]
    float yb[64];
#pragma unroll
    for (int q = 0; q < 16; ++q) {
        int idx = (q * 4 + half * 16) & 63;          // stays 16B-aligned
        float4 v = *(const float4*)&e2s[colb][idx];
        yb[q * 4 + 0] = v.x; yb[q * 4 + 1] = v.y;
        yb[q * 4 + 2] = v.z; yb[q * 4 + 3] = v.w;
    }

    const __bf16* A0 = Tt + (size_t)(0  + colw) * KTOT;
    const __bf16* A1 = Tt + (size_t)(16 + colw) * KTOT;
    const __bf16* A2 = Tt + (size_t)(32 + colw) * KTOT;
    const __bf16* A3 = Tt + (size_t)(48 + colw) * KTOT;

    v8f acc0 = {0.f,0.f,0.f,0.f,0.f,0.f,0.f,0.f};
    v8f acc1 = acc0, acc2 = acc0, acc3 = acc0;

    // ---- bilinear part: K = 0..4095; outer step 64 shares the e1 scalar ----
    for (int kk = 0; kk < KBI; kk += 64) {
        float a1 = e1s[colb][kk >> 6];               // 1 ds_load_b32 per 64 K
#pragma unroll
        for (int ss = 0; ss < 2; ++ss) {
            const int kx = kk + ss * 32;
            const int t0 = ss * 32;                  // compile-time reg base
            v16bf b;
            b[0]  = (__bf16)(a1 * yb[t0 + 0]);  b[1]  = (__bf16)(a1 * yb[t0 + 1]);
            b[2]  = (__bf16)(a1 * yb[t0 + 2]);  b[3]  = (__bf16)(a1 * yb[t0 + 3]);
            b[4]  = (__bf16)(a1 * yb[t0 + 4]);  b[5]  = (__bf16)(a1 * yb[t0 + 5]);
            b[6]  = (__bf16)(a1 * yb[t0 + 6]);  b[7]  = (__bf16)(a1 * yb[t0 + 7]);
            b[8]  = (__bf16)(a1 * yb[t0 + 8]);  b[9]  = (__bf16)(a1 * yb[t0 + 9]);
            b[10] = (__bf16)(a1 * yb[t0 + 10]); b[11] = (__bf16)(a1 * yb[t0 + 11]);
            b[12] = (__bf16)(a1 * yb[t0 + 12]); b[13] = (__bf16)(a1 * yb[t0 + 13]);
            b[14] = (__bf16)(a1 * yb[t0 + 14]); b[15] = (__bf16)(a1 * yb[t0 + 15]);

            acc0 = WMMA_BF16(load_afrag(A0, kx, half), b, acc0);
            acc1 = WMMA_BF16(load_afrag(A1, kx, half), b, acc1);
            acc2 = WMMA_BF16(load_afrag(A2, kx, half), b, acc2);
            acc3 = WMMA_BF16(load_afrag(A3, kx, half), b, acc3);
        }
    }

    // ---- feed-forward part: K = 4096..4223; B comes straight from LDS bf16 ----
#pragma unroll
    for (int kk = KBI; kk < KTOT; kk += 32) {
        int r0 = (kk - KBI) + half * 16;             // contiguous r per lane
        v8bf g0 = *(const v8bf*)&gsb[colb][r0];
        v8bf g1 = *(const v8bf*)&gsb[colb][r0 + 8];
        v16bf b = __builtin_shufflevector(g0, g1, 0, 1, 2, 3, 4, 5, 6, 7,
                                                  8, 9, 10, 11, 12, 13, 14, 15);
        acc0 = WMMA_BF16(load_afrag(A0, kk, half), b, acc0);
        acc1 = WMMA_BF16(load_afrag(A1, kk, half), b, acc1);
        acc2 = WMMA_BF16(load_afrag(A2, kk, half), b, acc2);
        acc3 = WMMA_BF16(load_afrag(A3, kk, half), b, acc3);
    }

    // ---- epilogue: D element v of m-tile m -> row m*16 + v + half*8 ----
    const size_t scol = (size_t)(s0 + colb);
#pragma unroll
    for (int v = 0; v < 8; ++v) {
        int r0 = v + half * 8;
        X[(size_t)(r0     ) * BSZ + scol] = tanhf(acc0[v] + bias[r0]);
        X[(size_t)(r0 + 16) * BSZ + scol] = tanhf(acc1[v] + bias[r0 + 16]);
        X[(size_t)(r0 + 32) * BSZ + scol] = tanhf(acc2[v] + bias[r0 + 32]);
        X[(size_t)(r0 + 48) * BSZ + scol] = tanhf(acc3[v] + bias[r0 + 48]);
    }
}

// ---------------------------------------------------------------------------
// kernel_launch: 1 prep + 5 fused-GEMM stages.
// (k,bs) -> (bs,d) reshapes are free reinterpretations; u/ur write d_out flat.
// ---------------------------------------------------------------------------
extern "C" void kernel_launch(void* const* d_in, const int* in_sizes, int n_in,
                              void* d_out, int out_size, void* d_ws, size_t ws_size,
                              hipStream_t stream) {
    const float* e1 = (const float*)d_in[0];
    const float* p  = (const float*)d_in[1];
    const float* e2 = (const float*)d_in[2];
    const float* er = (const float*)d_in[3];
    const float* T1 = (const float*)d_in[4];
    const float* T2 = (const float*)d_in[5];
    const float* T3 = (const float*)d_in[6];
    const float* W  = (const float*)d_in[7];
    const float* b  = (const float*)d_in[8];
    float* out = (float*)d_out;

    const size_t xsz = (size_t)DD * BSZ;           // 8.4M floats per intermediate
    float* Xa  = (float*)d_ws;
    float* Xc  = Xa + xsz;
    float* Xar = Xc + xsz;
    __bf16* Tt = (__bf16*)(Xar + xsz);             // 3 * 64*4224 bf16

    {
        int np = 3 * DD * KTOT;
        prep_weights<<<(np + 255) / 256, 256, 0, stream>>>(T1, T2, T3, W, Tt);
    }

    dim3 grid(BSZ / 64), block(128);
    const __bf16* Tt1 = Tt + 0 * (size_t)DD * KTOT;
    const __bf16* Tt2 = Tt + 1 * (size_t)DD * KTOT;
    const __bf16* Tt3 = Tt + 2 * (size_t)DD * KTOT;

    // a = ntn(e1, p, T1); c = ntn(p, e2, T2); ar = ntn(er, p, T1)
    ntn_stage<<<grid, block, 0, stream>>>(e1, p,  Tt1, b, Xa);
    ntn_stage<<<grid, block, 0, stream>>>(p,  e2, Tt2, b, Xc);
    ntn_stage<<<grid, block, 0, stream>>>(er, p,  Tt1, b, Xar);
    // u = ntn(a, c, T3); ur = ntn(ar, c, T3) -> straight into d_out (2,d,bs)
    ntn_stage<<<grid, block, 0, stream>>>(Xa,  Xc, Tt3, b, out);
    ntn_stage<<<grid, block, 0, stream>>>(Xar, Xc, Tt3, b, out + xsz);
}
